// ATTNetwork_4569845203602
// MI455X (gfx1250) — matrix operand order repaired and checked by reading, not compiled, not run on hardware
//
#include <hip/hip_runtime.h>
#include <hip/hip_bf16.h>
#include <math.h>

// ---------------------------------------------------------------------------
// MI455X (gfx1250) fused ATT-network.
//  Pass 1: column stats (mean/var) over 524288x256 inputs (coalesced, atomics)
//  Prep  : fold BN into fc1 weights; pre-swizzle W1/W2/W3 into bf16 hi/lo
//          WMMA B-fragment layout (per-lane contiguous 16 bf16 = 32B)
//  Pass 2: fused kernel, 128 rows (=16 batches of 8 agents) per workgroup:
//          fc1 via v_wmma_f32_16x16x32_bf16 with hi/lo split (3 WMMA/tile,
//          ~fp32 accuracy), fi kept in LDS, attention on VALU, fc2/fc3 via
//          WMMA (M=16 batches), q written straight to HBM.
// Assumes n_agents == 8 (reference setup) ; agent_i read from device scalar.
// ---------------------------------------------------------------------------

typedef __attribute__((ext_vector_type(16))) __bf16 v16bf;
typedef __attribute__((ext_vector_type(8)))  float  v8f;
typedef __attribute__((ext_vector_type(4)))  float  vf4;

#define D_IN      256
#define H_DIM     64
#define O_DIM     64
#define N_AG      8
#define BPB       16                  // batches per block
#define ROWS_PB   (BPB * N_AG)        // 128 rows per block
#define FI_STRIDE 68                  // padded (mult of 4 for b128, bank-skewed)
#define X2_STRIDE 132

// ws layout -----------------------------------------------------------------
#define WS_SUM    0                   // 256 floats
#define WS_SUMSQ  256                 // 256 floats
#define WS_B1P    512                 // 64 floats (folded fc1 bias)
#define WS_W1HI_B 4096                                // 8kc*4nt*32l*16 bf16
#define WS_W1LO_B (WS_W1HI_B + 8*4*32*16*2)
#define WS_W2HI_B (WS_W1LO_B + 8*4*32*16*2)           // 2kc*4nt*32l*16
#define WS_W2LO_B (WS_W2HI_B + 2*4*32*16*2)
#define WS_W3HI_B (WS_W2LO_B + 2*4*32*16*2)           // 4kc*4nt*32l*16
#define WS_W3LO_B (WS_W3HI_B + 4*4*32*16*2)
// total ~116 KB of ws

__device__ __forceinline__ v8f vzero8() {
  v8f z;
#pragma unroll
  for (int i = 0; i < 8; ++i) z[i] = 0.0f;
  return z;
}

// split fp32 -> bf16 hi + bf16 lo residual (Ozaki-style 2-term split)
__device__ __forceinline__ void split16(const float* a, v16bf& hi, v16bf& lo) {
#pragma unroll
  for (int i = 0; i < 16; ++i) {
    float v = a[i];
    __bf16 h = (__bf16)v;
    hi[i] = h;
    lo[i] = (__bf16)(v - (float)h);
  }
}

__device__ __forceinline__ v8f wmma_bf16(v16bf a, v16bf b, v8f c) {
  // (neg_a, A, neg_b, B, c_mod, C, reuse_a, reuse_b)
  return __builtin_amdgcn_wmma_f32_16x16x32_bf16(false, a, false, b,
                                                 (short)0, c, false, false);
}

// ---------------------------------------------------------------------------
__global__ void k_zero_stats(float* ws) {
  ws[threadIdx.x] = 0.0f;   // 512 threads: sum + sumsq
}

__global__ void k_col_stats(const float* __restrict__ in, float* __restrict__ ws,
                            int BN, int rowsPer) {
  const int col = threadIdx.x;                   // 256 threads = 256 columns
  const long r0 = (long)blockIdx.x * rowsPer;
  float s = 0.0f, s2 = 0.0f;
  for (int r = 0; r < rowsPer; ++r) {
    long row = r0 + r;
    if (row >= BN) break;
    float v = in[row * D_IN + col];
    s += v;
    s2 += v * v;
  }
  atomicAdd(&ws[WS_SUM + col], s);
  atomicAdd(&ws[WS_SUMSQ + col], s2);
}

// Fold BN into fc1; pre-swizzle all weight matrices into WMMA B-fragment
// layout: frag[kc][nt][lane][i] = W[h = nt*16 + lane%16][k = kc*32 + (lane/16)*8 + (i%8) + (i/8)*16]
__global__ void k_prep(const float* __restrict__ bn_w, const float* __restrict__ bn_b,
                       const float* __restrict__ fc1_w, const float* __restrict__ fc1_b,
                       const float* __restrict__ fc2_w, const float* __restrict__ fc3_w,
                       float* __restrict__ ws, int BN) {
  __shared__ float s_scale[D_IN];
  __shared__ float s_shift[D_IN];
  const int t = threadIdx.x;                     // 256 threads
  {
    float inv_n = 1.0f / (float)BN;
    float mean = ws[WS_SUM + t] * inv_n;
    float var  = ws[WS_SUMSQ + t] * inv_n - mean * mean;
    float sc   = bn_w[t] * rsqrtf(var + 1e-5f);
    s_scale[t] = sc;
    s_shift[t] = bn_b[t] - mean * sc;
  }
  __syncthreads();

  if (t < H_DIM) {                               // folded bias b1'
    float acc = fc1_b[t];
    for (int d = 0; d < D_IN; ++d) acc += fc1_w[t * D_IN + d] * s_shift[d];
    ws[WS_B1P + t] = acc;
  }

  __bf16* w1hi = (__bf16*)((char*)ws + WS_W1HI_B);
  __bf16* w1lo = (__bf16*)((char*)ws + WS_W1LO_B);
  for (int idx = t; idx < 8 * 4 * 32 * 16; idx += 256) {
    int i  = idx & 15;
    int ln = (idx >> 4) & 31;
    int nt = (idx >> 9) & 3;
    int kc = idx >> 11;
    int h  = nt * 16 + (ln & 15);
    int d  = kc * 32 + ((ln >> 4) << 3) + (i & 7) + ((i >> 3) << 4);
    float wv = fc1_w[h * D_IN + d] * s_scale[d];   // fold BN scale
    __bf16 hb = (__bf16)wv;
    w1hi[idx] = hb;
    w1lo[idx] = (__bf16)(wv - (float)hb);
  }

  __bf16* w2hi = (__bf16*)((char*)ws + WS_W2HI_B);
  __bf16* w2lo = (__bf16*)((char*)ws + WS_W2LO_B);
  for (int idx = t; idx < 2 * 4 * 32 * 16; idx += 256) {
    int i  = idx & 15;
    int ln = (idx >> 4) & 31;
    int nt = (idx >> 9) & 3;
    int kc = idx >> 11;
    int h  = nt * 16 + (ln & 15);
    int d  = kc * 32 + ((ln >> 4) << 3) + (i & 7) + ((i >> 3) << 4);
    float wv = fc2_w[h * H_DIM + d];
    __bf16 hb = (__bf16)wv;
    w2hi[idx] = hb;
    w2lo[idx] = (__bf16)(wv - (float)hb);
  }

  __bf16* w3hi = (__bf16*)((char*)ws + WS_W3HI_B);
  __bf16* w3lo = (__bf16*)((char*)ws + WS_W3LO_B);
  for (int idx = t; idx < 4 * 4 * 32 * 16; idx += 256) {
    int i  = idx & 15;
    int ln = (idx >> 4) & 31;
    int nt = (idx >> 9) & 3;
    int kc = idx >> 11;
    int h  = nt * 16 + (ln & 15);
    int d  = kc * 32 + ((ln >> 4) << 3) + (i & 7) + ((i >> 3) << 4);
    float wv = fc3_w[h * (2 * H_DIM) + d];
    __bf16 hb = (__bf16)wv;
    w3hi[idx] = hb;
    w3lo[idx] = (__bf16)(wv - (float)hb);
  }
}

// ---------------------------------------------------------------------------
// Fused main kernel: 256 threads (8 waves), 128 rows / 16 batches per block.
__global__ void __launch_bounds__(256, 2)
k_fused(const float* __restrict__ in, const float* __restrict__ ws,
        const float* __restrict__ corr, const float* __restrict__ fc2_b,
        const float* __restrict__ fc3_b, const int* __restrict__ agent_ip,
        float* __restrict__ out) {
  __shared__ __align__(16) float s_fi[ROWS_PB * FI_STRIDE];   // 34816 B
  __shared__ float s_corr[H_DIM * H_DIM];                     // 16384 B
  __shared__ float s_t[BPB * H_DIM];                          //  4096 B
  __shared__ __align__(16) float s_x2[BPB * X2_STRIDE];       //  8448 B
  __shared__ float s_beta[BPB * 8];                           //   512 B
  __shared__ float s_b1p[H_DIM];
  __shared__ float s_b2[H_DIM];
  __shared__ float s_b3[H_DIM];                               //   768 B

  const int tid  = threadIdx.x;
  const int wv   = tid >> 5;          // wave 0..7
  const int lane = tid & 31;
  const int m    = lane & 15;         // row-within-tile / batch index
  const int kb   = (lane >> 4) << 3;  // A-fragment K sub-offset {0,8}
  const int ai   = *agent_ip;

  // ---- stage constants into LDS -------------------------------------------
  for (int i = tid; i < H_DIM * H_DIM; i += 256) s_corr[i] = corr[i];
  if (tid < H_DIM) {
    s_b1p[tid] = ws[WS_B1P + tid];
    s_b2[tid]  = fc2_b[tid];
    s_b3[tid]  = fc3_b[tid];
  }
  __syncthreads();

  // ---- fc1: 16x64 output tile per wave, K=256, bf16 hi/lo split -----------
  const v16bf* w1hi = (const v16bf*)((const char*)ws + WS_W1HI_B);
  const v16bf* w1lo = (const v16bf*)((const char*)ws + WS_W1LO_B);

  const size_t grow = (size_t)blockIdx.x * ROWS_PB + wv * 16 + m;
  v8f acc[4] = {vzero8(), vzero8(), vzero8(), vzero8()};

#pragma unroll
  for (int kc = 0; kc < 8; ++kc) {
    const float* rp = in + grow * D_IN + kc * 32 + kb;
    float av[16];
    *(vf4*)&av[0]  = *(const vf4*)(rp);
    *(vf4*)&av[4]  = *(const vf4*)(rp + 4);
    *(vf4*)&av[8]  = *(const vf4*)(rp + 16);
    *(vf4*)&av[12] = *(const vf4*)(rp + 20);
    v16bf ahi, alo;
    split16(av, ahi, alo);
#pragma unroll
    for (int nt = 0; nt < 4; ++nt) {
      v16bf bhi = w1hi[(kc * 4 + nt) * 32 + lane];
      v16bf blo = w1lo[(kc * 4 + nt) * 32 + lane];
      acc[nt] = wmma_bf16(ahi, bhi, acc[nt]);
      acc[nt] = wmma_bf16(ahi, blo, acc[nt]);
      acc[nt] = wmma_bf16(alo, bhi, acc[nt]);
    }
  }
  // bias + relu -> fi in LDS (C layout: elem a -> M=a+(lane>=16?8:0), N=lane%16)
#pragma unroll
  for (int nt = 0; nt < 4; ++nt) {
    int col = nt * 16 + m;
    float b = s_b1p[col];
#pragma unroll
    for (int a = 0; a < 8; ++a) {
      int r = wv * 16 + a + ((lane >> 4) << 3);
      float v = acc[nt][a] + b;
      s_fi[r * FI_STRIDE + col] = v > 0.0f ? v : 0.0f;
    }
  }
  __syncthreads();

  // ---- waves 0-3: fc2 (h2) via WMMA ; waves 4-7: attention t = fa @ corr --
  if (wv < 4) {
    const v16bf* w2hi = (const v16bf*)((const char*)ws + WS_W2HI_B);
    const v16bf* w2lo = (const v16bf*)((const char*)ws + WS_W2LO_B);
    const int arow = m * N_AG + ai;             // fa row for batch m
    v8f c2 = vzero8();
#pragma unroll
    for (int kc = 0; kc < 2; ++kc) {
      const float* rp = &s_fi[arow * FI_STRIDE + kc * 32 + kb];
      float av[16];
      *(vf4*)&av[0]  = *(const vf4*)(rp);
      *(vf4*)&av[4]  = *(const vf4*)(rp + 4);
      *(vf4*)&av[8]  = *(const vf4*)(rp + 16);
      *(vf4*)&av[12] = *(const vf4*)(rp + 20);
      v16bf ahi, alo;
      split16(av, ahi, alo);
      v16bf bhi = w2hi[(kc * 4 + wv) * 32 + lane];
      v16bf blo = w2lo[(kc * 4 + wv) * 32 + lane];
      c2 = wmma_bf16(ahi, bhi, c2);
      c2 = wmma_bf16(ahi, blo, c2);
      c2 = wmma_bf16(alo, bhi, c2);
    }
    int n = wv * 16 + m;
    float b = s_b2[n];
#pragma unroll
    for (int a = 0; a < 8; ++a) {
      int bt = a + ((lane >> 4) << 3);
      float v = c2[a] + b;
      s_x2[bt * X2_STRIDE + n] = v > 0.0f ? v : 0.0f;   // h2 -> x2[:, 0:64]
    }
  } else {
    const int tid2 = tid - 128;
    const int b = tid2 >> 3, g = tid2 & 7;       // batch, 8-wide h group
    const int farow = b * N_AG + ai;
    float te[8] = {0, 0, 0, 0, 0, 0, 0, 0};
    for (int d = 0; d < H_DIM; ++d) {
      float fav = s_fi[farow * FI_STRIDE + d];
      const float* cr = &s_corr[d * H_DIM + g * 8];
#pragma unroll
      for (int i = 0; i < 8; ++i) te[i] += fav * cr[i];
    }
#pragma unroll
    for (int i = 0; i < 8; ++i) s_t[b * H_DIM + g * 8 + i] = te[i];
  }
  __syncthreads();

  // ---- beta_j = t . fj -----------------------------------------------------
  if (wv >= 4) {
    const int tid2 = tid - 128;
    const int b = tid2 >> 3, g = tid2 & 7;
    if (g < N_AG - 1) {
      int jr = g + (g >= ai ? 1 : 0);
      const float* tv = &s_t[b * H_DIM];
      const float* fj = &s_fi[(b * N_AG + jr) * FI_STRIDE];
      float s = 0.0f;
      for (int e = 0; e < H_DIM; ++e) s += tv[e] * fj[e];
      s_beta[b * 8 + g] = s;
    }
  }
  __syncthreads();

  // ---- softmax + vi -> x2[:, 64:128] --------------------------------------
  if (wv >= 4) {
    const int tid2 = tid - 128;
    const int b = tid2 >> 3, g = tid2 & 7;
    float bj[N_AG - 1];
    float mx = -3.0e38f;
#pragma unroll
    for (int j = 0; j < N_AG - 1; ++j) {
      bj[j] = s_beta[b * 8 + j];
      mx = bj[j] > mx ? bj[j] : mx;
    }
    float sum = 0.0f;
#pragma unroll
    for (int j = 0; j < N_AG - 1; ++j) {
      bj[j] = __expf(bj[j] - mx);
      sum += bj[j];
    }
    float inv = 1.0f / sum;
    float vi[8] = {0, 0, 0, 0, 0, 0, 0, 0};
#pragma unroll
    for (int j = 0; j < N_AG - 1; ++j) {
      int jr = j + (j >= ai ? 1 : 0);
      float al = bj[j] * inv;
      const float* fj = &s_fi[(b * N_AG + jr) * FI_STRIDE + g * 8];
#pragma unroll
      for (int i = 0; i < 8; ++i) vi[i] += al * fj[i];
    }
#pragma unroll
    for (int i = 0; i < 8; ++i)
      s_x2[b * X2_STRIDE + H_DIM + g * 8 + i] = vi[i];
  }
  __syncthreads();

  // ---- fc3: q = [h2|vi] @ W3^T + b3 via WMMA (waves 0-3), write to HBM ----
  if (wv < 4) {
    const v16bf* w3hi = (const v16bf*)((const char*)ws + WS_W3HI_B);
    const v16bf* w3lo = (const v16bf*)((const char*)ws + WS_W3LO_B);
    v8f c3 = vzero8();
#pragma unroll
    for (int kc = 0; kc < 4; ++kc) {
      const float* rp = &s_x2[m * X2_STRIDE + kc * 32 + kb];
      float av[16];
      *(vf4*)&av[0]  = *(const vf4*)(rp);
      *(vf4*)&av[4]  = *(const vf4*)(rp + 4);
      *(vf4*)&av[8]  = *(const vf4*)(rp + 16);
      *(vf4*)&av[12] = *(const vf4*)(rp + 20);
      v16bf ahi, alo;
      split16(av, ahi, alo);
      v16bf bhi = w3hi[(kc * 4 + wv) * 32 + lane];
      v16bf blo = w3lo[(kc * 4 + wv) * 32 + lane];
      c3 = wmma_bf16(ahi, bhi, c3);
      c3 = wmma_bf16(ahi, blo, c3);
      c3 = wmma_bf16(alo, bhi, c3);
    }
    int n = wv * 16 + m;
    float b = s_b3[n];
#pragma unroll
    for (int a = 0; a < 8; ++a) {
      int bt = a + ((lane >> 4) << 3);
      out[((size_t)blockIdx.x * BPB + bt) * O_DIM + n] = c3[a] + b;
    }
  }
}

// ---------------------------------------------------------------------------
extern "C" void kernel_launch(void* const* d_in, const int* in_sizes, int n_in,
                              void* d_out, int out_size, void* d_ws, size_t ws_size,
                              hipStream_t stream) {
  (void)n_in; (void)out_size; (void)ws_size;
  const float* inputs = (const float*)d_in[0];
  const float* bn_w   = (const float*)d_in[1];
  const float* bn_b   = (const float*)d_in[2];
  const float* fc1_w  = (const float*)d_in[3];
  const float* fc1_b  = (const float*)d_in[4];
  const float* fc2_w  = (const float*)d_in[5];
  const float* fc2_b  = (const float*)d_in[6];
  const float* fc3_w  = (const float*)d_in[7];
  const float* fc3_b  = (const float*)d_in[8];
  const float* corr   = (const float*)d_in[9];
  const int*  agent_i = (const int*)d_in[11];
  float* out = (float*)d_out;
  float* ws  = (float*)d_ws;

  const int BN = in_sizes[0] / D_IN;              // 524288 rows

  k_zero_stats<<<1, 512, 0, stream>>>(ws);

  const int rowsPer = 512;
  const int nblk = (BN + rowsPer - 1) / rowsPer;
  k_col_stats<<<nblk, 256, 0, stream>>>(inputs, ws, BN, rowsPer);

  k_prep<<<1, 256, 0, stream>>>(bn_w, bn_b, fc1_w, fc1_b, fc2_w, fc3_w, ws, BN);

  k_fused<<<BN / ROWS_PB, 256, 0, stream>>>(inputs, ws, corr, fc2_b, fc3_b,
                                            agent_i, out);
}
